// VectorQuantizer_15710990369630
// MI455X (gfx1250) — matrix operand order repaired and checked by reading, not compiled, not run on hardware
//
#include <hip/hip_runtime.h>

#define BATCH    131072
#define EMBED    64
#define NCODES   1024
#define NTILES   (NCODES / 16)                        // 64 code tiles
#define ROW_TILES 2
#define WAVES_PER_BLOCK 8
#define ROWS_PER_WAVE  (ROW_TILES * 16)               // 32
#define ROWS_PER_BLOCK (WAVES_PER_BLOCK * ROWS_PER_WAVE) // 256
#define NBLOCKS  (BATCH / ROWS_PER_BLOCK)             // 512

typedef __attribute__((ext_vector_type(16))) __bf16 v16bf;
typedef __attribute__((ext_vector_type(8)))  float  v8f;

struct B32 { uint4 a, b; };                           // 32 bytes

__device__ __forceinline__ v16bf join16(uint4 a, uint4 b) {
  B32 t{a, b};
  return __builtin_bit_cast(v16bf, t);
}

__device__ __forceinline__ v8f wmma_bf16(v16bf a, v16bf b, v8f c) {
  // D = A(16x32 bf16) * B(32x16 bf16) + C(16x16 f32)
  return __builtin_amdgcn_wmma_f32_16x16x32_bf16(
      /*neg_a=*/false, a, /*neg_b=*/false, b,
      /*c_mod=*/(short)0, c, /*reuse_a=*/false, /*reuse_b=*/false);
}

__device__ __forceinline__ void split16(const float* vals, v16bf& hi, v16bf& lo) {
#pragma unroll
  for (int e = 0; e < 16; ++e) {
    __bf16 h = (__bf16)vals[e];
    hi[e] = h;
    lo[e] = (__bf16)(vals[e] - (float)h);
  }
}

// A fragment (16x32, MxK): lane = 16*h + m.
// element e of v16bf holds K = kk*32 + 8*h + (e<8 ? e : e+8)
__device__ __forceinline__ void load_a_frag(const float* __restrict__ zrow,
                                            int h, int kk, v16bf& hi, v16bf& lo) {
  const int d0 = kk * 32 + 8 * h;
  float4 f0 = *(const float4*)(zrow + d0);
  float4 f1 = *(const float4*)(zrow + d0 + 4);
  float4 f2 = *(const float4*)(zrow + d0 + 16);
  float4 f3 = *(const float4*)(zrow + d0 + 20);
  float v[16] = {f0.x, f0.y, f0.z, f0.w, f1.x, f1.y, f1.z, f1.w,
                 f2.x, f2.y, f2.z, f2.w, f3.x, f3.y, f3.z, f3.w};
  split16(v, hi, lo);
}

// ---------------- prep: build pre-swizzled bf16 hi/lo codebook fragments,
// ---------------- fp32 code norms, zero hist + loss --------------------------
// bfrag layout (uint4 units): [ct][part 0..7][lane 0..31]
//   part 0/1 = hi chunk0 (e0-7 / e8-15), part 2/3 = hi chunk1,
//   part 4/5 = lo chunk0,                part 6/7 = lo chunk1.
// B fragment (32x16 KxN): lane = 16*h + n; element e holds K = kk*32 + 8*h + (e<8 ? e : e+8)
__global__ __launch_bounds__(256) void vq_prep(const float* __restrict__ cb,
                                               uint4* __restrict__ bfrag,
                                               float* __restrict__ cnorm,
                                               unsigned* __restrict__ hist,
                                               float* __restrict__ loss_acc) {
  int tid = blockIdx.x * 256 + threadIdx.x;           // 0..2047
  if (tid < NCODES) {
    const float4* p = (const float4*)(cb + (size_t)tid * EMBED);
    float s = 0.f;
#pragma unroll
    for (int i = 0; i < EMBED / 4; ++i) {
      float4 v = p[i];
      s = fmaf(v.x, v.x, fmaf(v.y, v.y, fmaf(v.z, v.z, fmaf(v.w, v.w, s))));
    }
    cnorm[tid] = s;
    hist[tid] = 0u;
  }
  if (tid == 0) *loss_acc = 0.f;

  // one thread per (code tile, lane): builds that lane's 4 v16bf fragments
  int ct = tid >> 5;
  int lane = tid & 31;
  int h = lane >> 4;
  int col = lane & 15;
  const float* crow = cb + (size_t)(ct * 16 + col) * EMBED;
  uint4* base = bfrag + (size_t)ct * (8 * 32);
#pragma unroll
  for (int kk = 0; kk < 2; ++kk) {
    const int d0 = kk * 32 + 8 * h;
    float4 f0 = *(const float4*)(crow + d0);
    float4 f1 = *(const float4*)(crow + d0 + 4);
    float4 f2 = *(const float4*)(crow + d0 + 16);
    float4 f3 = *(const float4*)(crow + d0 + 20);
    float v[16] = {f0.x, f0.y, f0.z, f0.w, f1.x, f1.y, f1.z, f1.w,
                   f2.x, f2.y, f2.z, f2.w, f3.x, f3.y, f3.z, f3.w};
    v16bf hi, lo;
    split16(v, hi, lo);
    B32 th = __builtin_bit_cast(B32, hi);
    B32 tl = __builtin_bit_cast(B32, lo);
    base[(2 * kk + 0) * 32 + lane] = th.a;
    base[(2 * kk + 1) * 32 + lane] = th.b;
    base[(4 + 2 * kk + 0) * 32 + lane] = tl.a;
    base[(4 + 2 * kk + 1) * 32 + lane] = tl.b;
  }
}

// ---------------- main: WMMA distances + fused argmin + gather ---------------
__global__ __launch_bounds__(256) void vq_main(const float* __restrict__ z_e,
                                               const float* __restrict__ cb,
                                               const uint4* __restrict__ bfrag,
                                               const float* __restrict__ cnorm,
                                               float* __restrict__ out_zq,
                                               float* __restrict__ out_idx,
                                               unsigned* __restrict__ hist,
                                               float* __restrict__ loss_acc) {
  const int lane = threadIdx.x & 31;
  const int wave = threadIdx.x >> 5;
  const int h = lane >> 4;       // half-wave selector
  const int col = lane & 15;     // column / row-in-tile selector
  const int rowbase0 = (blockIdx.x * WAVES_PER_BLOCK + wave) * ROWS_PER_WAVE;

  // Register-resident A fragments (hi/lo bf16 split) for ROW_TILES row tiles.
  v16bf Ahi[ROW_TILES][2], Alo[ROW_TILES][2];
#pragma unroll
  for (int t = 0; t < ROW_TILES; ++t) {
    const float* zrow = z_e + (size_t)(rowbase0 + t * 16 + col) * EMBED;
#pragma unroll
    for (int kk = 0; kk < 2; ++kk)
      load_a_frag(zrow, h, kk, Ahi[t][kk], Alo[t][kk]);
  }

  float minv[ROW_TILES][8];
  int   mini[ROW_TILES][8];
#pragma unroll
  for (int t = 0; t < ROW_TILES; ++t)
#pragma unroll
    for (int v = 0; v < 8; ++v) { minv[t][v] = 3.402823466e38f; mini[t][v] = 0; }

  const uint4* bfl = bfrag + lane;

  auto load_tile = [&](uint4* dst, int ct_) {
    const uint4* p = bfl + (size_t)ct_ * (8 * 32);
#pragma unroll
    for (int q = 0; q < 8; ++q) dst[q] = p[q * 32];
  };

  auto compute_tile = [&](const uint4* nb, float cn, int code) {
    v16bf Bhi0 = join16(nb[0], nb[1]);
    v16bf Bhi1 = join16(nb[2], nb[3]);
    v16bf Blo0 = join16(nb[4], nb[5]);
    v16bf Blo1 = join16(nb[6], nb[7]);
#pragma unroll
    for (int t = 0; t < ROW_TILES; ++t) {
      v8f acc = {};
      // bf16x3: hi*hi + hi*lo + lo*hi, fp32 accumulate, K=64 in 2 chunks
      acc = wmma_bf16(Ahi[t][0], Bhi0, acc);
      acc = wmma_bf16(Ahi[t][1], Bhi1, acc);
      acc = wmma_bf16(Ahi[t][0], Blo0, acc);
      acc = wmma_bf16(Ahi[t][1], Blo1, acc);
      acc = wmma_bf16(Alo[t][0], Bhi0, acc);
      acc = wmma_bf16(Alo[t][1], Bhi1, acc);
#pragma unroll
      for (int v = 0; v < 8; ++v) {
        float s = fmaf(-2.f, acc[v], cn);   // ||c||^2 - 2 z.c (argmin-equivalent)
        if (s < minv[t][v]) { minv[t][v] = s; mini[t][v] = code; }
      }
    }
  };

  // 2x unrolled ping-pong pipeline: loads for one buffer overlap the other
  // buffer's 12 WMMAs; no register rotation, no WAR hazards on B operands.
  uint4 b0[8], b1[8];
  load_tile(b0, 0);
  float cn0 = cnorm[col];
  for (int ct = 0; ct < NTILES; ct += 2) {
    const int ct2 = (ct + 2 < NTILES) ? (ct + 2) : 0;  // clamped (branch-free)
    load_tile(b1, ct + 1);
    float cn1 = cnorm[(ct + 1) * 16 + col];
    compute_tile(b0, cn0, ct * 16 + col);
    load_tile(b0, ct2);
    cn0 = cnorm[ct2 * 16 + col];
    compute_tile(b1, cn1, (ct + 1) * 16 + col);
  }

  // Reduce across the 16 columns (lanes 0-15 own rows 0-7, lanes 16-31 rows 8-15)
  float lsum = 0.f;
#pragma unroll
  for (int t = 0; t < ROW_TILES; ++t) {
    const int rowbase = rowbase0 + t * 16;
#pragma unroll
    for (int v = 0; v < 8; ++v) {
#pragma unroll
      for (int s = 8; s >= 1; s >>= 1) {
        float ov = __shfl_xor(minv[t][v], s, 32);
        int   oi = __shfl_xor(mini[t][v], s, 32);
        if (ov < minv[t][v] || (ov == minv[t][v] && oi < mini[t][v])) {
          minv[t][v] = ov; mini[t][v] = oi;
        }
      }
    }
    if (col == 0) {  // lane 0 -> rows 0..7, lane 16 -> rows 8..15
#pragma unroll
      for (int v = 0; v < 8; ++v) {
        int row = rowbase + v + 8 * h;
        int idx = mini[t][v];
        out_idx[row] = (float)idx;
        atomicAdd(&hist[idx], 1u);
      }
    }
    // Gather z_q, write straight-through output, accumulate squared diffs.
#pragma unroll
    for (int r = 0; r < 16; ++r) {
      int src = (r < 8) ? 0 : 16;
      int idx = __shfl(mini[t][r & 7], src, 32);
      int row = rowbase + r;
      float2 q = ((const float2*)(cb + (size_t)idx * EMBED))[lane];
      float2 z = ((const float2*)(z_e + (size_t)row * EMBED))[lane];
      float d0 = q.x - z.x, d1 = q.y - z.y;
      float2 o; o.x = z.x + d0; o.y = z.y + d1;  // z_e + stopgrad(z_q - z_e)
      ((float2*)(out_zq + (size_t)row * EMBED))[lane] = o;
      lsum = fmaf(d0, d0, fmaf(d1, d1, lsum));
    }
  }
#pragma unroll
  for (int s = 16; s >= 1; s >>= 1) lsum += __shfl_xor(lsum, s, 32);
  if (lane == 0) atomicAdd(loss_acc, lsum);
}

// ---------------- finalize: losses + perplexity ------------------------------
__global__ __launch_bounds__(1024) void vq_fin(const unsigned* __restrict__ hist,
                                               const float* __restrict__ loss_acc,
                                               float* __restrict__ out3) {
  __shared__ float sm[1024];
  int i = threadIdx.x;
  float p = (float)hist[i] * (1.0f / (float)BATCH);
  sm[i] = p * logf(p + 1e-10f);
  __syncthreads();
#pragma unroll
  for (int s = 512; s > 0; s >>= 1) {
    if (i < s) sm[i] += sm[i + s];
    __syncthreads();
  }
  if (i == 0) {
    float perp = expf(-sm[0]);
    float cl = loss_acc[0] * (1.0f / ((float)BATCH * (float)EMBED));
    out3[0] = cl;            // codebook_loss
    out3[1] = 0.25f * cl;    // commitment_loss
    out3[2] = perp;          // perplexity
  }
}

extern "C" void kernel_launch(void* const* d_in, const int* in_sizes, int n_in,
                              void* d_out, int out_size, void* d_ws, size_t ws_size,
                              hipStream_t stream) {
  const float* z_e = (const float*)d_in[0];   // [131072, 64] f32
  const float* cb  = (const float*)d_in[1];   // [1024, 64]   f32

  float* out      = (float*)d_out;
  float* out_zq   = out;                             // [131072*64]
  float* out_idx  = out + (size_t)BATCH * EMBED;     // [131072] (as float)
  float* out_loss = out_idx + BATCH;                 // [3]

  // ws layout: bfrag (256KB, 16B aligned) | cnorm (4KB) | hist (4KB) | loss (4B)
  uint4*    bfrag    = (uint4*)d_ws;                       // 64*8*32 uint4
  float*    cnorm    = (float*)(bfrag + NTILES * 8 * 32);  // 1024 f32
  unsigned* hist     = (unsigned*)(cnorm + NCODES);        // 1024 u32
  float*    loss_acc = (float*)(hist + NCODES);            // 1 f32

  vq_prep<<<8, 256, 0, stream>>>(cb, bfrag, cnorm, hist, loss_acc);
  vq_main<<<NBLOCKS, 256, 0, stream>>>(z_e, cb, bfrag, cnorm, out_zq, out_idx,
                                       hist, loss_acc);
  vq_fin<<<1, 1024, 0, stream>>>(hist, loss_acc, out_loss);
}
